// SelfAttention_53644141527559
// MI455X (gfx1250) — matrix operand order repaired and checked by reading, not compiled, not run on hardware
//
#include <hip/hip_runtime.h>
#include <hip/hip_bf16.h>

#define DEVINL __device__ __forceinline__

typedef __attribute__((ext_vector_type(16))) __bf16          v16bf;
typedef __attribute__((ext_vector_type(8)))  float           v8f;
typedef __attribute__((ext_vector_type(16))) unsigned short  ushort16;
typedef __attribute__((ext_vector_type(4)))  unsigned int    v4u;
typedef __attribute__((ext_vector_type(8)))  int             v8i;
typedef __attribute__((ext_vector_type(4)))  int             v4i;

constexpr int kB  = 2;
constexpr int kS  = 2048;
constexpr int kH  = 16;
constexpr int kD  = 64;     // head dim
constexpr int kDM = 1024;   // d_model

#if defined(__has_builtin)
#if __has_builtin(__builtin_amdgcn_tensor_load_to_lds) && \
    __has_builtin(__builtin_amdgcn_s_wait_tensorcnt)
#define HAS_TDM 1
#endif
#endif
#ifndef HAS_TDM
#define HAS_TDM 0
#endif

// ---------- bf16 helpers ----------
DEVINL unsigned short f32_to_bf16(float f) {
  unsigned int u = __builtin_bit_cast(unsigned int, f);
  u += 0x7FFFu + ((u >> 16) & 1u);       // round-to-nearest-even
  return (unsigned short)(u >> 16);
}

DEVINL v8f wmma_bf16(v16bf a, v16bf b, v8f c) {
  // (neg_a, A, neg_b, B, c_mod, C, reuse_a, reuse_b)
  return __builtin_amdgcn_wmma_f32_16x16x32_bf16(false, a, false, b, (short)0, c,
                                                 false, false);
}

// A-fragment (16x32 bf16): lane = 16*h2 + m.
// element j in [0,16) -> K = kbase + 16*h2 + j
DEVINL v16bf load_a_frag(const unsigned short* rowptr, int kbase, int h2) {
  union { uint4 v; unsigned short s[8]; } c0, c1;
  c0.v = *reinterpret_cast<const uint4*>(rowptr + kbase + 16 * h2);
  c1.v = *reinterpret_cast<const uint4*>(rowptr + kbase + 16 * h2 + 8);
  ushort16 u;
#pragma unroll
  for (int j = 0; j < 8; ++j) { u[j] = c0.s[j]; u[8 + j] = c1.s[j]; }
  return __builtin_bit_cast(v16bf, u);
}

DEVINL v16bf load_a_frag_f32(const float* rowptr, int kbase, int h2) {
  ushort16 u;
#pragma unroll
  for (int j = 0; j < 8; ++j) u[j]     = f32_to_bf16(rowptr[kbase + 16 * h2 + j]);
#pragma unroll
  for (int j = 0; j < 8; ++j) u[8 + j] = f32_to_bf16(rowptr[kbase + 16 * h2 + 8 + j]);
  return __builtin_bit_cast(v16bf, u);
}

// B-fragment (32x16 bf16): lane = 16*h2 + n; element j -> K = base + 16*h2 + j
// (caller folds 16*h2 into the pointer; 16 contiguous bf16 = 32 bytes)
DEVINL v16bf load_b_frag(const unsigned short* p16) {
  union { uint4 v[2]; ushort16 u; } c;
  const uint4* q = reinterpret_cast<const uint4*>(p16);
  c.v[0] = q[0];
  c.v[1] = q[1];
  return __builtin_bit_cast(v16bf, c.u);
}

DEVINL v16bf load_b_frag_f32(const float* p) {
  ushort16 u;
#pragma unroll
  for (int j = 0; j < 16; ++j) u[j] = f32_to_bf16(p[j]);
  return __builtin_bit_cast(v16bf, u);
}

// ---------- Tensor Data Mover: 2-D tile -> LDS (D# per CDNA5 ISA) ----------
// data_size = 2 bytes (bf16). tile/tensor dims and stride in elements.
DEVINL void tdm_load_tile_2d(const unsigned short* gsrc, unsigned int lds_byte,
                             unsigned int tile0, unsigned int tile1,
                             unsigned long long stride0) {
#if HAS_TDM
  const unsigned long long ga = (unsigned long long)gsrc;
  v4u g0;
  g0[0] = 1u;                                            // count=1, user D#
  g0[1] = lds_byte;                                      // lds_addr (bytes)
  g0[2] = (unsigned int)(ga & 0xFFFFFFFFull);            // global_addr[31:0]
  g0[3] = (unsigned int)((ga >> 32) & 0x1FFFFFFull)      // global_addr[56:32]
          | (2u << 30);                                  // type = 2 ("image")
  v8i g1;
  g1[0] = (int)(1u << 16);                               // data_size=1 (2B), wg_mask=0
  g1[1] = (int)((tile0 & 0xFFFFu) << 16);                // tensor_dim0[15:0]
  g1[2] = (int)(((tile0 >> 16) & 0xFFFFu)                // tensor_dim0[31:16]
          | ((tile1 & 0xFFFFu) << 16));                  // tensor_dim1[15:0]
  g1[3] = (int)(((tile1 >> 16) & 0xFFFFu)                // tensor_dim1[31:16]
          | ((tile0 & 0xFFFFu) << 16));                  // tile_dim0
  g1[4] = (int)(tile1 & 0xFFFFu);                        // tile_dim1 (tile_dim2=0)
  g1[5] = (int)(stride0 & 0xFFFFFFFFull);                // tensor_dim0_stride[31:0]
  g1[6] = (int)((stride0 >> 32) & 0xFFFFull);            // stride[47:32] (dim1_stride=0)
  g1[7] = 0;
  const v4i z4 = {0, 0, 0, 0};                           // groups 2/3 unused (2-D)
  const v8i z8 = {0, 0, 0, 0, 0, 0, 0, 0};
  __builtin_amdgcn_tensor_load_to_lds(g0, g1, z4, z4, z8, 0);
#else
  (void)gsrc; (void)lds_byte; (void)tile0; (void)tile1; (void)stride0;
#endif
}

// ===================================================================
// Kernel 1: per-head QKV projections.
// grid = B*H*(S/16), block = 96 (3 waves: wave0->Q, wave1->K, wave2->V^T)
// Qh,Kh: (B,H,S,64) bf16 ; Vt: (B,H,64,S) bf16 (transposed for PV B-frags)
// ===================================================================
__global__ void __launch_bounds__(96)
qkv_proj_kernel(const float* __restrict__ q_in, const float* __restrict__ k_in,
                const float* __restrict__ v_in,
                const float* __restrict__ Wq, const float* __restrict__ bq,
                const float* __restrict__ Wk, const float* __restrict__ bk,
                const float* __restrict__ Wv, const float* __restrict__ bv,
                unsigned short* __restrict__ Qh, unsigned short* __restrict__ Kh,
                unsigned short* __restrict__ Vt) {
  const int mt = blockIdx.x % (kS / 16);
  const int bh = blockIdx.x / (kS / 16);
  const int h  = bh % kH;
  const int b  = bh / kH;
  const int wave = threadIdx.x >> 5;
  const int lane = threadIdx.x & 31;
  const int h2 = lane >> 4;
  const int ln = lane & 15;
  const int t0 = mt * 16;

  const float* X    = (wave == 0) ? q_in : (wave == 1) ? k_in : v_in;
  const float* W    = (wave == 0) ? Wq   : (wave == 1) ? Wk   : Wv;
  const float* bias = (wave == 0) ? bq   : (wave == 1) ? bk   : bv;

  // A: token rows, input head-dim cols.  X layout: (B,S,DM), head slice h*64.
  const float* xrow = X + ((size_t)(b * kS + t0 + ln)) * kDM + h * kD;
  v16bf a0 = load_a_frag_f32(xrow, 0, h2);
  v16bf a1 = load_a_frag_f32(xrow, 32, h2);

#pragma unroll
  for (int nt = 0; nt < 4; ++nt) {
    const int n0 = nt * 16;
    v8f c = {};
    // B(k,n) = W[n, k]  (out = x @ W^T): lane's row of W is contiguous in k.
    v16bf b0 = load_b_frag_f32(W + (size_t)(n0 + ln) * kD + 0  + 16 * h2);
    v16bf b1 = load_b_frag_f32(W + (size_t)(n0 + ln) * kD + 32 + 16 * h2);
    c = wmma_bf16(a0, b0, c);
    c = wmma_bf16(a1, b1, c);

    const float bv_ = bias[n0 + ln];
#pragma unroll
    for (int r = 0; r < 8; ++r) {
      const int m = r + 8 * h2;          // C layout: VGPR r -> M = r + 8*half
      const int t = t0 + m;
      const unsigned short val = f32_to_bf16(c[r] + bv_);
      if (wave == 0) {
        Qh[((size_t)bh * kS + t) * kD + n0 + ln] = val;
      } else if (wave == 1) {
        Kh[((size_t)bh * kS + t) * kD + n0 + ln] = val;
      } else {
        Vt[((size_t)bh * kD + n0 + ln) * kS + t] = val;   // transposed
      }
    }
  }
}

// ===================================================================
// Kernel 2: fused flash attention per (b,h).
// grid = B*H*(S/64), block = 128 (4 waves, 16 q-rows each)
// K/V tiles double-buffered in LDS via TDM (wave 0 drives the DMA).
// Ocat: (B,S,DM) bf16, heads concatenated at col h*64+d.
// ===================================================================
// LDS map (ushort elements):
//   Kbuf[buf] @ buf*2048          : 32 keys x 64 d   (row-major d)
//   Vbuf[buf] @ 4096 + buf*2048   : 64 d   x 32 keys (row-major keys)
//   Ptile[w]  @ 8192 + w*512      : 16 x 32 bf16 P staging per wave
__global__ void __launch_bounds__(128)
attn_kernel(const unsigned short* __restrict__ Qh,
            const unsigned short* __restrict__ Kh,
            const unsigned short* __restrict__ Vt,
            const int* __restrict__ mask,
            unsigned short* __restrict__ Ocat) {
  __shared__ __align__(16) unsigned short smem[2 * 2048 + 2 * 2048 + 4 * 512];

  const int nqb  = kS / 64;
  const int bh   = blockIdx.x / nqb;
  const int qb   = blockIdx.x % nqb;
  const int b    = bh / kH;
  const int h    = bh % kH;
  const int wave = threadIdx.x >> 5;
  const int lane = threadIdx.x & 31;
  const int tid  = threadIdx.x;
  const int h2   = lane >> 4;
  const int ln   = lane & 15;
  const int q0   = qb * 64 + wave * 16;

  const unsigned short* Qbase = Qh + (size_t)bh * kS * kD;
  const unsigned short* Kbase = Kh + (size_t)bh * kS * kD;
  const unsigned short* Vbase = Vt + (size_t)bh * kD * kS;
  const int* mbase = mask + (size_t)b * kS * kS;

  unsigned short* Kbuf0 = smem;
  unsigned short* Vbuf0 = smem + 4096;
  unsigned short* pl    = smem + 8192 + wave * 512;

  // ---- tile staging: TDM (wave 0) or cooperative copy fallback ----
  auto stage = [&](int buf, int kb) {
#if HAS_TDM
    if (wave == 0) {
      const unsigned int koff = (unsigned int)(size_t)(Kbuf0 + buf * 2048);
      const unsigned int voff = (unsigned int)(size_t)(Vbuf0 + buf * 2048);
      // K tile: 32 rows (keys) x 64 contiguous d, row stride 64 elems.
      tdm_load_tile_2d(Kbase + (size_t)kb * kD, koff, 64u, 32u, 64ull);
      // V tile: 64 rows (d) x 32 contiguous keys, row stride S elems.
      tdm_load_tile_2d(Vbase + kb, voff, 32u, 64u, (unsigned long long)kS);
    }
#else
    unsigned short* dk = Kbuf0 + buf * 2048;
    unsigned short* dv = Vbuf0 + buf * 2048;
    const unsigned short* sk = Kbase + (size_t)kb * kD;   // contiguous 2048
    const unsigned short* sv = Vbase + kb;                // 64 rows, stride kS
#pragma unroll 2
    for (int i = tid; i < 256; i += 128)
      reinterpret_cast<uint4*>(dk)[i] = reinterpret_cast<const uint4*>(sk)[i];
#pragma unroll 2
    for (int i = tid; i < 256; i += 128) {
      const int row = i >> 2, ch = i & 3;
      *reinterpret_cast<uint4*>(dv + row * 32 + ch * 8) =
          *reinterpret_cast<const uint4*>(sv + (size_t)row * kS + ch * 8);
    }
#endif
  };

  // Q A-fragments resident for whole K sweep.
  const unsigned short* qrow = Qbase + (size_t)(q0 + ln) * kD;
  v16bf qa0 = load_a_frag(qrow, 0, h2);
  v16bf qa1 = load_a_frag(qrow, 32, h2);

  v8f o0 = {}, o1 = {}, o2 = {}, o3 = {};
  float rmax[8], rsum[8];
#pragma unroll
  for (int r = 0; r < 8; ++r) { rmax[r] = -3.4e38f; rsum[r] = 0.0f; }

  const float scale  = 0.03125f;              // 1/sqrt(d_model) = 1/32
  const float NEGBIG = -1e20f * 0.03125f;     // masked energy after scaling

  stage(0, 0);                                // prologue DMA into buffer 0

  for (int kb = 0; kb < kS; kb += 32) {
    const int cur = (kb >> 5) & 1;
#if HAS_TDM
    if (wave == 0) __builtin_amdgcn_s_wait_tensorcnt(0);
#endif
    __syncthreads();                          // buffer[cur] visible to all waves
    if (kb + 32 < kS) stage(cur ^ 1, kb + 32);

    // streaming hint for next mask tile (global_prefetch_b8)
    __builtin_prefetch(mbase + (size_t)(q0 + ln) * kS + kb + 32 + 16 * h2, 0, 0);

    const unsigned short* kt = Kbuf0 + cur * 2048;   // [key][d]
    const unsigned short* vt = Vbuf0 + cur * 2048;   // [d][key]

    // ---- S = Q * K^T for two 16-key tiles (4 WMMA, B-frags from LDS) ----
    v8f s0 = {}, s1 = {};
    {
      const unsigned short* krow = kt + (size_t)ln * kD + 16 * h2;
      s0 = wmma_bf16(qa0, load_b_frag(krow), s0);
      s0 = wmma_bf16(qa1, load_b_frag(krow + 32), s0);
    }
    {
      const unsigned short* krow = kt + (size_t)(16 + ln) * kD + 16 * h2;
      s1 = wmma_bf16(qa0, load_b_frag(krow), s1);
      s1 = wmma_bf16(qa1, load_b_frag(krow + 32), s1);
    }

    // ---- mask + scale ----
#pragma unroll
    for (int r = 0; r < 8; ++r) {
      const int qr = q0 + r + 8 * h2;
      const int m0 = mbase[(size_t)qr * kS + kb + ln];
      const int m1 = mbase[(size_t)qr * kS + kb + 16 + ln];
      s0[r] = m0 ? s0[r] * scale : NEGBIG;
      s1[r] = m1 ? s1[r] * scale : NEGBIG;
    }

    // ---- online softmax (row reductions across 16-lane halves) ----
#pragma unroll
    for (int r = 0; r < 8; ++r) {
      float t = fmaxf(s0[r], s1[r]);
#pragma unroll
      for (int xm = 1; xm <= 8; xm <<= 1) t = fmaxf(t, __shfl_xor(t, xm, 32));
      const float mn   = fmaxf(rmax[r], t);
      const float corr = __expf(rmax[r] - mn);
      const float p0   = __expf(s0[r] - mn);
      const float p1   = __expf(s1[r] - mn);
      s0[r] = p0; s1[r] = p1;
      float ts = p0 + p1;
#pragma unroll
      for (int xm = 1; xm <= 8; xm <<= 1) ts += __shfl_xor(ts, xm, 32);
      rsum[r] = rsum[r] * corr + ts;
      rmax[r] = mn;
      o0[r] *= corr; o1[r] *= corr; o2[r] *= corr; o3[r] *= corr;
    }

    // ---- C-layout -> A-layout via wave-private LDS tile (16x32 bf16) ----
#pragma unroll
    for (int r = 0; r < 8; ++r) {
      const int row = r + 8 * h2;
      pl[row * 32 + ln]      = f32_to_bf16(s0[r]);
      pl[row * 32 + 16 + ln] = f32_to_bf16(s1[r]);
    }
    __syncthreads();
    const v16bf pa = load_a_frag(pl + ln * 32, 0, h2);
    __syncthreads();

    // ---- O += P * V  (4 WMMA over d-tiles; V^T B-frags from LDS) ----
    {
      const unsigned short* v0 = vt + (size_t)(0 * 16 + ln) * 32 + 16 * h2;
      const unsigned short* v1 = vt + (size_t)(1 * 16 + ln) * 32 + 16 * h2;
      const unsigned short* v2 = vt + (size_t)(2 * 16 + ln) * 32 + 16 * h2;
      const unsigned short* v3 = vt + (size_t)(3 * 16 + ln) * 32 + 16 * h2;
      o0 = wmma_bf16(pa, load_b_frag(v0), o0);
      o1 = wmma_bf16(pa, load_b_frag(v1), o1);
      o2 = wmma_bf16(pa, load_b_frag(v2), o2);
      o3 = wmma_bf16(pa, load_b_frag(v3), o3);
    }
  }

  // ---- epilogue: normalize & store concatenated heads ----
#pragma unroll
  for (int r = 0; r < 8; ++r) {
    const int row = q0 + r + 8 * h2;
    const float inv = 1.0f / rsum[r];
    const size_t base = ((size_t)(b * kS + row)) * kDM + h * kD;
    Ocat[base + 0 * 16 + ln] = f32_to_bf16(o0[r] * inv);
    Ocat[base + 1 * 16 + ln] = f32_to_bf16(o1[r] * inv);
    Ocat[base + 2 * 16 + ln] = f32_to_bf16(o2[r] * inv);
    Ocat[base + 3 * 16 + ln] = f32_to_bf16(o3[r] * inv);
  }
}

// ===================================================================
// Kernel 3: output projection  out = Ocat @ Wo^T + bo  (f32 out)
// grid = (TOK/16, DM/64), block = 128 (4 waves, one 16-col tile each)
// ===================================================================
__global__ void __launch_bounds__(128)
out_proj_kernel(const unsigned short* __restrict__ Ocat,
                const float* __restrict__ Wo, const float* __restrict__ bo,
                float* __restrict__ out) {
  const int wave = threadIdx.x >> 5;
  const int lane = threadIdx.x & 31;
  const int h2 = lane >> 4;
  const int ln = lane & 15;
  const int t0 = blockIdx.x * 16;
  const int n0 = blockIdx.y * 64 + wave * 16;

  const unsigned short* arow = Ocat + (size_t)(t0 + ln) * kDM;
  const float*          wrow = Wo + (size_t)(n0 + ln) * kDM;

  v8f c = {};
  for (int k0 = 0; k0 < kDM; k0 += 32) {
    __builtin_prefetch(wrow + k0 + 128, 0, 0);
    v16bf a = load_a_frag(arow, k0, h2);
    v16bf w = load_b_frag_f32(wrow + k0 + 16 * h2);
    c = wmma_bf16(a, w, c);
  }

  const float bias = bo[n0 + ln];
#pragma unroll
  for (int r = 0; r < 8; ++r) {
    const int t = t0 + r + 8 * h2;
    out[(size_t)t * kDM + n0 + ln] = c[r] + bias;
  }
}

// ===================================================================
extern "C" void kernel_launch(void* const* d_in, const int* in_sizes, int n_in,
                              void* d_out, int out_size, void* d_ws, size_t ws_size,
                              hipStream_t stream) {
  const float* values = (const float*)d_in[0];
  const float* keys   = (const float*)d_in[1];
  const float* query  = (const float*)d_in[2];
  const int*   mask   = (const int*)  d_in[3];
  const float* Wq = (const float*)d_in[4];
  const float* bq = (const float*)d_in[5];
  const float* Wk = (const float*)d_in[6];
  const float* bk = (const float*)d_in[7];
  const float* Wv = (const float*)d_in[8];
  const float* bv = (const float*)d_in[9];
  const float* Wo = (const float*)d_in[10];
  const float* bo = (const float*)d_in[11];
  float* out = (float*)d_out;

  // Workspace layout (bf16 halves): Qh | Kh | Vt | Ocat  (8 MB each, 32 MB total)
  const size_t QKV = (size_t)kB * kH * kS * kD;   // 4M elements
  unsigned short* Qh   = (unsigned short*)d_ws;
  unsigned short* Kh   = Qh + QKV;
  unsigned short* Vt   = Kh + QKV;
  unsigned short* Ocat = Vt + QKV;                // (B,S,DM) = 4M elements

  (void)in_sizes; (void)n_in; (void)out_size; (void)ws_size;

  qkv_proj_kernel<<<kB * kH * (kS / 16), 96, 0, stream>>>(
      query, keys, values, Wq, bq, Wk, bk, Wv, bv, Qh, Kh, Vt);

  attn_kernel<<<kB * kH * (kS / 64), 128, 0, stream>>>(Qh, Kh, Vt, mask, Ocat);

  out_proj_kernel<<<dim3(kB * kS / 16, kDM / 64), 128, 0, stream>>>(Ocat, Wo, bo, out);
}
